// TrainingFreeAttentionBlocks_74938589381053
// MI455X (gfx1250) — matrix operand order repaired and checked by reading, not compiled, not run on hardware
//
#include <hip/hip_runtime.h>
#include <hip/hip_bf16.h>
#include <math.h>

// ---- CDNA5 WMMA types --------------------------------------------------
typedef __attribute__((ext_vector_type(16))) __bf16 v16bf;
typedef __attribute__((ext_vector_type(8)))  float  v8f;
typedef int v4i __attribute__((vector_size(16)));

union BF16x16 { v16bf v; uint4 q[2]; };

#define NPIX 9216      // 96*96
#define CDIM 1280
#define HA   96
#define NK   2304      // 48*48
#define TRI  589824    // 768*768
#define MATN 5242880   // 1280*64*64

#define KSTEP 32
#define MT    128                 // macro-tile rows (i)
#define NT    64                  // macro-tile cols (j)
#define SQ_ELEMS (MT * KSTEP)     // 4096 bf16 per buffer
#define SK_ELEMS (NT * KSTEP)     // 2048 bf16 per buffer

// ---- optional CDNA5 async global->LDS path ------------------------------
#if defined(__has_builtin)
#if __has_builtin(__builtin_amdgcn_global_load_async_to_lds_b128) && \
    __has_builtin(__builtin_amdgcn_s_wait_asynccnt)
#define USE_ASYNC 1
#else
#define USE_ASYNC 0
#endif
#else
#define USE_ASYNC 0
#endif

#if USE_ASYNC
__device__ __forceinline__ void async_ld_b128(const void* g, void* l) {
  // (global src, LDS dst, imm offset, imm cpol); params are generic v4i*.
  __builtin_amdgcn_global_load_async_to_lds_b128((v4i*)g, (v4i*)l, 0, 0);
}
#endif

// ---- Stage 1: mask = 8x8 min-pool of (guidance > 0.5); zero attn_out ----
__global__ void k_mask(const float* __restrict__ guidance,
                       float* __restrict__ m, float* __restrict__ attn_out) {
  int i = blockIdx.x * blockDim.x + threadIdx.x;
  if (i >= NPIX) return;
  int y = i / HA, x = i % HA;
  const float* g = guidance + (size_t)y * 8 * 768 + x * 8;
  float mn = 1.0f;
  for (int dy = 0; dy < 8; ++dy)
    for (int dx = 0; dx < 8; ++dx)
      mn = fminf(mn, g[dy * 768 + dx] > 0.5f ? 1.0f : 0.0f);
  m[i] = mn;
  attn_out[i] = 0.0f;
}

// ---- Stage 2: bicubic 64->96 (Keys a=-0.5, half-pixel), pack bf16 -------
__device__ __forceinline__ float keys_w(float x) {
  x = fabsf(x);
  if (x < 1.0f) return ((1.5f * x - 2.5f) * x) * x + 1.0f;
  if (x < 2.0f) return ((-0.5f * x + 2.5f) * x - 4.0f) * x + 2.0f;
  return 0.0f;
}

__global__ void k_resize(const float* __restrict__ ref, const float* __restrict__ cor,
                         const float* __restrict__ m,
                         __bf16* __restrict__ Qb, __bf16* __restrict__ Kb) {
  int i = blockIdx.x * blockDim.x + threadIdx.x;
  if (i >= NPIX) return;
  int c = blockIdx.y;
  int oy = i / HA, ox = i % HA;
  const float scale = 2.0f / 3.0f;
  float fy = (oy + 0.5f) * scale - 0.5f;
  float fx = (ox + 0.5f) * scale - 0.5f;
  int by = (int)floorf(fy), bx = (int)floorf(fx);
  float ty = fy - by, tx = fx - bx;
  float wy[4], wx[4]; int iy[4], ix[4];
  float sy = 0.0f, sx = 0.0f;
  for (int t = 0; t < 4; ++t) {
    wy[t] = keys_w(ty - (float)(t - 1)); sy += wy[t];
    wx[t] = keys_w(tx - (float)(t - 1)); sx += wx[t];
    iy[t] = min(max(by + t - 1, 0), 63);
    ix[t] = min(max(bx + t - 1, 0), 63);
  }
  float inv = 1.0f / (sy * sx);
  const float* rch = ref + (size_t)c * 4096;
  const float* cch = cor + (size_t)c * 4096;
  float ar = 0.0f, ac = 0.0f;
  for (int t = 0; t < 4; ++t) {
    const float* r0 = rch + iy[t] * 64;
    const float* c0 = cch + iy[t] * 64;
    float rr = 0.0f, rc = 0.0f;
    for (int u = 0; u < 4; ++u) { rr += wx[u] * r0[ix[u]]; rc += wx[u] * c0[ix[u]]; }
    ar += wy[t] * rr; ac += wy[t] * rc;
  }
  ar *= inv; ac *= inv;
  Qb[(size_t)i * CDIM + c] = (__bf16)(ar * m[i]);  // q = ref * m
  Kb[(size_t)i * CDIM + c] = (__bf16)ac;           // k = feat
}

// ---- core: 128x64 macro-tile GEMM, double-buffered K staging -----------
// Wave (miw in 0..3, njw in 0..1) owns a 2x2 grid of 16x16 f32 accumulators.
__device__ __forceinline__ void gemm128x64(
    const __bf16* __restrict__ Qb, const __bf16* __restrict__ Kb,
    __bf16* sQ, __bf16* sK,            // [2*SQ_ELEMS], [2*SK_ELEMS]
    int i0, int j0, int tid, int miw, int njw, int l15, int half,
    v8f acc[2][2]) {
  // staging map: 256 threads cover Q 128x32 (2x16B each) and K 64x32 (1x16B)
  const int qrow = tid >> 1;               // 0..127
  const int qcp  = (tid & 1) * 16;         // bf16 offset: 0 or 16
  const int krow = tid >> 2;               // 0..63
  const int kch  = (tid & 3) * 8;

  const __bf16* gq = &Qb[(size_t)(i0 + qrow) * CDIM + qcp];
  const __bf16* gk = &Kb[(size_t)(j0 + krow) * CDIM + kch];
  const int lqo = qrow * KSTEP + qcp;
  const int lko = krow * KSTEP + kch;

  // prologue: stage chunk 0 into buffer 0
#if USE_ASYNC
  async_ld_b128(gq,     &sQ[lqo]);
  async_ld_b128(gq + 8, &sQ[lqo + 8]);
  async_ld_b128(gk,     &sK[lko]);
  __builtin_amdgcn_s_wait_asynccnt(0);
#else
  {
    uint4 rq0 = *(const uint4*)gq;
    uint4 rq1 = *(const uint4*)(gq + 8);
    uint4 rk0 = *(const uint4*)gk;
    *(uint4*)&sQ[lqo] = rq0;
    *(uint4*)&sQ[lqo + 8] = rq1;
    *(uint4*)&sK[lko] = rk0;
  }
#endif
  __syncthreads();

  int p = 0;
  for (int kk = 0; kk < CDIM; kk += KSTEP) {
    const int nb = p ^ 1;
    const bool last = (kk + KSTEP >= CDIM);

    // issue next-chunk loads early so latency hides behind WMMA
#if USE_ASYNC
    if (!last) {
      async_ld_b128(gq + kk + KSTEP,     &sQ[nb * SQ_ELEMS + lqo]);
      async_ld_b128(gq + kk + KSTEP + 8, &sQ[nb * SQ_ELEMS + lqo + 8]);
      async_ld_b128(gk + kk + KSTEP,     &sK[nb * SK_ELEMS + lko]);
    }
#else
    uint4 rq0, rq1, rk0;
    if (!last) {
      rq0 = *(const uint4*)(gq + kk + KSTEP);
      rq1 = *(const uint4*)(gq + kk + KSTEP + 8);
      rk0 = *(const uint4*)(gk + kk + KSTEP);
    }
#endif

    // compute from buffer p: 4 WMMAs per wave per stage
    {
      const __bf16* sQp = &sQ[p * SQ_ELEMS];
      const __bf16* sKp = &sK[p * SK_ELEMS];
      BF16x16 A[2], B[2];
      for (int a = 0; a < 2; ++a) {
        const __bf16* qp = &sQp[(miw * 32 + a * 16 + l15) * KSTEP];
        A[a].q[0] = *(const uint4*)(qp + half * 8);
        A[a].q[1] = *(const uint4*)(qp + 16 + half * 8);
      }
      for (int b = 0; b < 2; ++b) {
        const __bf16* kp = &sKp[(njw * 32 + b * 16 + l15) * KSTEP];
        B[b].q[0] = *(const uint4*)(kp + half * 8);
        B[b].q[1] = *(const uint4*)(kp + 16 + half * 8);
      }
      acc[0][0] = __builtin_amdgcn_wmma_f32_16x16x32_bf16(false, A[0].v, false, B[0].v, (short)0, acc[0][0], false, false);
      acc[0][1] = __builtin_amdgcn_wmma_f32_16x16x32_bf16(false, A[0].v, false, B[1].v, (short)0, acc[0][1], false, false);
      acc[1][0] = __builtin_amdgcn_wmma_f32_16x16x32_bf16(false, A[1].v, false, B[0].v, (short)0, acc[1][0], false, false);
      acc[1][1] = __builtin_amdgcn_wmma_f32_16x16x32_bf16(false, A[1].v, false, B[1].v, (short)0, acc[1][1], false, false);
    }

    // commit next chunk into the other buffer
#if USE_ASYNC
    if (!last) __builtin_amdgcn_s_wait_asynccnt(0);
#else
    if (!last) {
      *(uint4*)&sQ[nb * SQ_ELEMS + lqo] = rq0;
      *(uint4*)&sQ[nb * SQ_ELEMS + lqo + 8] = rq1;
      *(uint4*)&sK[nb * SK_ELEMS + lko] = rk0;
    }
#endif
    __syncthreads();
    p = nb;
  }
}

// ---- Stage 3a: WMMA sweep 1 — online per-row (max,sum) stats ------------
__global__ void __launch_bounds__(256) k_attn_pass1(
    const __bf16* __restrict__ Qb, const __bf16* __restrict__ Kb,
    float* __restrict__ rowMp, float* __restrict__ rowLp) {
  __shared__ __align__(16) __bf16 sQ[2 * SQ_ELEMS];
  __shared__ __align__(16) __bf16 sK[2 * SK_ELEMS];
  const int i0 = blockIdx.x * MT;
  const int js = blockIdx.y;                  // j-split 0..7
  const int tid = threadIdx.x;
  const int lane = tid & 31;
  const int wid = tid >> 5;
  const int miw = wid & 3, njw = wid >> 2;
  const int l15 = lane & 15, half = lane >> 4;

  float rm[2][8], rl[2][8];
  for (int a = 0; a < 2; ++a)
    for (int r = 0; r < 8; ++r) { rm[a][r] = -3.0e38f; rl[a][r] = 0.0f; }

  for (int t = 0; t < 18; ++t) {
    const int j0 = js * 1152 + t * NT;
    v8f acc[2][2] = {};
    gemm128x64(Qb, Kb, sQ, sK, i0, j0, tid, miw, njw, l15, half, acc);

    for (int a = 0; a < 2; ++a)
      for (int r = 0; r < 8; ++r) {
        float v0 = acc[a][0][r] * 1.0e-3f;
        float v1 = acc[a][1][r] * 1.0e-3f;
        float mx = fmaxf(v0, v1);
        for (int o = 1; o < 16; o <<= 1) mx = fmaxf(mx, __shfl_xor(mx, o, 32));
        float sm = __expf(v0 - mx) + __expf(v1 - mx);
        for (int o = 1; o < 16; o <<= 1) sm += __shfl_xor(sm, o, 32);
        float nM = fmaxf(rm[a][r], mx);
        rl[a][r] = rl[a][r] * __expf(rm[a][r] - nM) + sm * __expf(mx - nM);
        rm[a][r] = nM;
      }
  }
  if (l15 == 0) {
    const int slab = js * 2 + njw;
    for (int a = 0; a < 2; ++a)
      for (int r = 0; r < 8; ++r) {
        int i = i0 + miw * 32 + a * 16 + half * 8 + r;
        rowMp[slab * NPIX + i] = rm[a][r];
        rowLp[slab * NPIX + i] = rl[a][r];
      }
  }
}

// ---- Stage 3b: combine 16 partial (M,L) slabs per row -------------------
__global__ void k_combine(const float* __restrict__ rowMp, const float* __restrict__ rowLp,
                          const float* __restrict__ m,
                          float* __restrict__ rowM, float* __restrict__ wrow) {
  int i = blockIdx.x * blockDim.x + threadIdx.x;
  if (i >= NPIX) return;
  float M = -3.0e38f;
  for (int s = 0; s < 16; ++s) M = fmaxf(M, rowMp[s * NPIX + i]);
  float L = 0.0f;
  for (int s = 0; s < 16; ++s) L += rowLp[s * NPIX + i] * __expf(rowMp[s * NPIX + i] - M);
  rowM[i] = M;
  wrow[i] = (m[i] > 0.0f && L > 0.0f) ? (1.0f / L) : 0.0f;  // ind[i]/L[i]
}

// ---- Stage 3c: WMMA sweep 2 — p = exp(s-M)/L * ind, reduce over i -------
__global__ void __launch_bounds__(256) k_attn_pass2(
    const __bf16* __restrict__ Qb, const __bf16* __restrict__ Kb,
    const float* __restrict__ rowM, const float* __restrict__ wrow,
    float* __restrict__ attn_out) {
  __shared__ __align__(16) __bf16 sQ[2 * SQ_ELEMS];
  __shared__ __align__(16) __bf16 sK[2 * SK_ELEMS];
  const int i0 = blockIdx.x * MT;
  const int js = blockIdx.y;
  const int tid = threadIdx.x;
  const int lane = tid & 31;
  const int wid = tid >> 5;
  const int miw = wid & 3, njw = wid >> 2;
  const int l15 = lane & 15, half = lane >> 4;

  float rmv[2][8], wr[2][8];
  for (int a = 0; a < 2; ++a)
    for (int r = 0; r < 8; ++r) {
      int i = i0 + miw * 32 + a * 16 + half * 8 + r;
      rmv[a][r] = rowM[i];
      wr[a][r] = wrow[i];
    }

  for (int t = 0; t < 18; ++t) {
    const int j0 = js * 1152 + t * NT;
    v8f acc[2][2] = {};
    gemm128x64(Qb, Kb, sQ, sK, i0, j0, tid, miw, njw, l15, half, acc);

    float ps0 = 0.0f, ps1 = 0.0f;
    for (int a = 0; a < 2; ++a)
      for (int r = 0; r < 8; ++r) {
        ps0 += __expf(acc[a][0][r] * 1.0e-3f - rmv[a][r]) * wr[a][r];
        ps1 += __expf(acc[a][1][r] * 1.0e-3f - rmv[a][r]) * wr[a][r];
      }
    ps0 += __shfl_xor(ps0, 16, 32);   // fold the two M-halves of each column
    ps1 += __shfl_xor(ps1, 16, 32);
    if (lane < 16) {
      atomicAdd(&attn_out[j0 + njw * 32 + lane], ps0);
      atomicAdd(&attn_out[j0 + njw * 32 + 16 + lane], ps1);
    }
  }
}

// ---- Stage 4a: x1 = bilinear(align_corners) 96x96 -> 48x48 --------------
__global__ void k_x1(const float* __restrict__ attn_out, float* __restrict__ xout) {
  int t = blockIdx.x * blockDim.x + threadIdx.x;
  if (t >= NK) return;
  int oy = t / 48, ox = t % 48;
  float fy = oy * (95.0f / 47.0f), fx = ox * (95.0f / 47.0f);
  int y0 = (int)floorf(fy), x0 = (int)floorf(fx);
  int y1 = min(y0 + 1, 95), x1 = min(x0 + 1, 95);
  float wy = fy - y0, wx = fx - x0;
  float a = attn_out[y0 * 96 + x0], b = attn_out[y0 * 96 + x1];
  float c = attn_out[y1 * 96 + x0], d = attn_out[y1 * 96 + x1];
  xout[t] = (a * (1.0f - wx) + b * wx) * (1.0f - wy) + (c * (1.0f - wx) + d * wx) * wy;
}

// ---- Stage 4b: out48[j] = sum_k A[j,k] * x1[k] --------------------------
__global__ void k_matvec(const float* __restrict__ A, const float* __restrict__ xv,
                         float* __restrict__ out48) {
  __shared__ float red[256];
  int j = blockIdx.x;
  float s = 0.0f;
  for (int k = threadIdx.x; k < NK; k += 256)
    s += A[(size_t)j * NK + k] * xv[k];
  red[threadIdx.x] = s;
  __syncthreads();
  for (int o = 128; o > 0; o >>= 1) {
    if (threadIdx.x < o) red[threadIdx.x] += red[threadIdx.x + o];
    __syncthreads();
  }
  if (threadIdx.x == 0) out48[j] = red[0];
}

// ---- Stage 5: trimap (48->768 bilinear) | ft_matting copy | mask_out ----
__global__ void k_final(const float* __restrict__ out48, const float* __restrict__ matt,
                        const float* __restrict__ attn_out, float* __restrict__ out) {
  int t = blockIdx.x * blockDim.x + threadIdx.x;
  if (t < TRI) {
    int oy = t / 768, ox = t % 768;
    float fy = (oy + 0.5f) * (48.0f / 768.0f) - 0.5f;
    float fx = (ox + 0.5f) * (48.0f / 768.0f) - 0.5f;
    int y0 = (int)floorf(fy), x0 = (int)floorf(fx);
    float wy = fy - y0, wx = fx - x0;
    int iy0 = min(max(y0, 0), 47), iy1 = min(max(y0 + 1, 0), 47);
    int ix0 = min(max(x0, 0), 47), ix1 = min(max(x0 + 1, 0), 47);
    float a = out48[iy0 * 48 + ix0], b = out48[iy0 * 48 + ix1];
    float c = out48[iy1 * 48 + ix0], d = out48[iy1 * 48 + ix1];
    out[t] = (a * (1.0f - wx) + b * wx) * (1.0f - wy) + (c * (1.0f - wx) + d * wx) * wy;
  } else if (t < TRI + MATN) {
    out[t] = matt[t - TRI];
  } else if (t < TRI + MATN + NPIX) {
    out[t] = attn_out[t - TRI - MATN];
  }
}

// ---- Launch -------------------------------------------------------------
extern "C" void kernel_launch(void* const* d_in, const int* in_sizes, int n_in,
                              void* d_out, int out_size, void* d_ws, size_t ws_size,
                              hipStream_t stream) {
  const float* ref  = (const float*)d_in[0];  // feature_of_reference_image
  const float* cor  = (const float*)d_in[1];  // ft_cor
  const float* amap = (const float*)d_in[2];  // attn_maps [2304,2304]
  const float* matt = (const float*)d_in[3];  // ft_matting (passthrough)
  const float* guid = (const float*)d_in[4];  // guidance
  float* out = (float*)d_out;

  char* ws = (char*)d_ws;
  float*  m        = (float*)(ws + 0);                    // 9216 f32
  float*  attn_out = (float*)(ws + 36864);                // 9216 f32
  float*  rowM     = (float*)(ws + 73728);                // 9216 f32
  float*  wrow     = (float*)(ws + 110592);               // 9216 f32
  float*  x1       = (float*)(ws + 147456);               // 2304 f32
  float*  out48    = (float*)(ws + 156672);               // 2304 f32
  float*  rowMp    = (float*)(ws + 165888);               // 16*9216 f32
  float*  rowLp    = (float*)(ws + 755712);               // 16*9216 f32
  __bf16* Qb       = (__bf16*)(ws + 1345536);             // 9216*1280 bf16
  __bf16* Kb       = (__bf16*)(ws + 1345536 + 23592960);  // 9216*1280 bf16

  k_mask   <<<36, 256, 0, stream>>>(guid, m, attn_out);
  k_resize <<<dim3(36, 1280), 256, 0, stream>>>(ref, cor, m, Qb, Kb);
  k_attn_pass1<<<dim3(72, 8), 256, 0, stream>>>(Qb, Kb, rowMp, rowLp);
  k_combine<<<36, 256, 0, stream>>>(rowMp, rowLp, m, rowM, wrow);
  k_attn_pass2<<<dim3(72, 8), 256, 0, stream>>>(Qb, Kb, rowM, wrow, attn_out);
  k_x1     <<<9, 256, 0, stream>>>(attn_out, x1);
  k_matvec <<<2304, 256, 0, stream>>>(amap, x1, out48);
  k_final  <<<22820, 256, 0, stream>>>(out48, matt, attn_out, out);
}
